// VN_Py_LSTM_20607253086888
// MI455X (gfx1250) — matrix operand order repaired and checked by reading, not compiled, not run on hardware
//
#include <hip/hip_runtime.h>

typedef __bf16 bf16;
typedef __attribute__((ext_vector_type(16))) __bf16 bf16x16;
typedef __attribute__((ext_vector_type(8)))  __bf16 bf16x8;
typedef __attribute__((ext_vector_type(8)))  float  f32x8;

#define T_LEN 512
#define BATCH 32
#define HID   512
#define G4H   2048  // 4*HID

// ---------------- CDNA5 wave32 WMMA fragment loaders (ISA 7.12.2) ----------------

// A-matrix 16x32 bf16 tile. base -> element (row0,k0) of tile, ld = row stride (elems).
// lanes 0-15: row m, K 0..15 in elems 0-15
// lanes 16-31: row m, K 16..31
__device__ __forceinline__ bf16x16 load_frag_a(const bf16* base, int ld, int lane) {
  const int m  = lane & 15;
  const int hi = lane >> 4;
  const bf16* row = base + m * ld;
  bf16x8 lo = *reinterpret_cast<const bf16x8*>(row + (hi ? 16 : 0));
  bf16x8 up = *reinterpret_cast<const bf16x8*>(row + (hi ? 24 : 8));
  bf16x16 r;
#pragma unroll
  for (int i = 0; i < 8; ++i) { r[i] = lo[i]; r[i + 8] = up[i]; }
  return r;
}

// B-matrix 32x16 bf16 tile sourced from row-major [N,K] weights (B column n == W row n).
// wbase -> W[n0*ld + k0]. lane n (0-15): K 0..15; lane n+16: K 16..31.
__device__ __forceinline__ bf16x16 load_frag_b(const bf16* wbase, int ld, int lane) {
  const int n    = lane & 15;
  const int koff = (lane >> 4) ? 16 : 0;
  const bf16* p = wbase + n * ld + koff;
  bf16x8 lo = *reinterpret_cast<const bf16x8*>(p);
  bf16x8 up = *reinterpret_cast<const bf16x8*>(p + 8);
  bf16x16 r;
#pragma unroll
  for (int i = 0; i < 8; ++i) { r[i] = lo[i]; r[i + 8] = up[i]; }
  return r;
}

__device__ __forceinline__ f32x8 wmma_bf16(bf16x16 a, bf16x16 b, f32x8 c) {
  return __builtin_amdgcn_wmma_f32_16x16x32_bf16(false, a, false, b, (short)0, c,
                                                 false, false);
}

__device__ __forceinline__ float sigmoid_(float x) {
  return __builtin_amdgcn_rcpf(1.0f + __expf(-x));
}
__device__ __forceinline__ float tanh_(float x) {
  float e = __expf(2.0f * x);
  return (e - 1.0f) * __builtin_amdgcn_rcpf(e + 1.0f);
}

// ---------------- fp32 -> bf16 conversion ----------------
__global__ void cvt_f32_bf16(const float* __restrict__ in, bf16* __restrict__ out, int n) {
  int i = blockIdx.x * blockDim.x + threadIdx.x;
  if (i < n) out[i] = (bf16)in[i];
}

// ---------------- input projection: xproj[M,4H] = X[M,H] @ Wih^T + (bih+bhh) ----------------
// grid (M/128, 4H/128), 256 threads (8 waves, 2x4). Wave tile 64(M) x 32(N).
__global__ void lstm_gemm_xproj(const bf16* __restrict__ X,    // [M, HID] bf16
                                const bf16* __restrict__ Wih,  // [4H, HID] bf16
                                const float* __restrict__ bih, // [4H]
                                const float* __restrict__ bhh, // [4H]
                                float* __restrict__ xproj)     // [M, 4H]
{
  const int lane  = threadIdx.x & 31;
  const int wave  = threadIdx.x >> 5;
  const int waveM = wave >> 2;  // 0..1
  const int waveN = wave & 3;   // 0..3
  const int m_base = blockIdx.x * 128 + waveM * 64;
  const int n_base = blockIdx.y * 128 + waveN * 32;

  f32x8 acc[4][2];
#pragma unroll
  for (int i = 0; i < 4; ++i)
#pragma unroll
    for (int j = 0; j < 2; ++j) acc[i][j] = (f32x8){0.f,0.f,0.f,0.f,0.f,0.f,0.f,0.f};

  for (int k = 0; k < HID; k += 32) {
    // Unconditional speculative prefetch of next K chunk (wrapped in-bounds: no branch).
    const int kp = (k + 32) & (HID - 1);
    __builtin_prefetch(Wih + (size_t)(n_base + (lane & 31)) * HID + kp, 0, 1);

    bf16x16 a[4], b[2];
#pragma unroll
    for (int i = 0; i < 4; ++i)
      a[i] = load_frag_a(X + (size_t)(m_base + 16 * i) * HID + k, HID, lane);
#pragma unroll
    for (int j = 0; j < 2; ++j)
      b[j] = load_frag_b(Wih + (size_t)(n_base + 16 * j) * HID + k, HID, lane);
#pragma unroll
    for (int i = 0; i < 4; ++i)
#pragma unroll
      for (int j = 0; j < 2; ++j) acc[i][j] = wmma_bf16(a[i], b[j], acc[i][j]);
  }

  const int m_lane = lane & 15;
  const int hi = lane >> 4;
#pragma unroll
  for (int j = 0; j < 2; ++j) {
    const int col = n_base + 16 * j + m_lane;
    const float bias = bih[col] + bhh[col];
#pragma unroll
    for (int i = 0; i < 4; ++i) {
#pragma unroll
      for (int e = 0; e < 8; ++e) {
        const int row = m_base + 16 * i + hi * 8 + e;
        xproj[(size_t)row * G4H + col] = acc[i][j][e] + bias;
      }
    }
  }
}

// ---------------- recurrent scan: one persistent workgroup, 32 waves ----------------
// Wave w owns hidden columns j0=w*16..+15 for ALL FOUR gates, so i/f/g/o of a given
// (batch, j) sit in the same lane+slot and the cell state c stays in registers.
__global__ void lstm_scan(const float* __restrict__ xproj,  // [T, B, 4H] fp32
                          const bf16*  __restrict__ Whh,    // [4H, HID] bf16
                          bf16*  __restrict__ out_bf,       // [T, B, HID]
                          float* __restrict__ out_f,        // [T, B, HID]
                          int write_bf, int write_f)
{
  __shared__ bf16 hsh[BATCH][HID + 8];  // +8 bf16 pad -> 4-bank row shift, conflict-free b128
  const int tid  = threadIdx.x;
  const int lane = tid & 31;
  const int wave = tid >> 5;

  for (int i = tid; i < BATCH * (HID + 8); i += 1024) (&hsh[0][0])[i] = (bf16)0.f;
  __syncthreads();

  const int j0     = wave * 16;
  const int m_lane = lane & 15;
  const int hi     = lane >> 4;

  f32x8 c0 = (f32x8){0.f,0.f,0.f,0.f,0.f,0.f,0.f,0.f};
  f32x8 c1 = c0;

  for (int t = 0; t < T_LEN; ++t) {
    const float* xp = xproj + (size_t)t * BATCH * G4H;

    // z accumulators seeded with the precomputed input projection
    f32x8 z[4][2];
#pragma unroll
    for (int g = 0; g < 4; ++g) {
      const int col = g * HID + j0 + m_lane;
#pragma unroll
      for (int mt = 0; mt < 2; ++mt)
#pragma unroll
        for (int e = 0; e < 8; ++e)
          z[g][mt][e] = xp[(size_t)(mt * 16 + hi * 8 + e) * G4H + col];
    }

    // z += h_{t-1} @ Whh^T  (A from LDS, B streamed from L2-resident weights)
    for (int k = 0; k < HID; k += 32) {
      const int kp = (k + 32) & (HID - 1);  // wrapped next-chunk prefetch, branch-free
      bf16x16 a0 = load_frag_a(&hsh[0][0]  + k, HID + 8, lane);  // batch rows 0-15
      bf16x16 a1 = load_frag_a(&hsh[16][0] + k, HID + 8, lane);  // batch rows 16-31
#pragma unroll
      for (int g = 0; g < 4; ++g) {
        __builtin_prefetch(Whh + (size_t)(g * HID + j0 + m_lane) * HID + kp, 0, 1);
        bf16x16 bb = load_frag_b(Whh + (size_t)(g * HID + j0) * HID + k, HID, lane);
        z[g][0] = wmma_bf16(a0, bb, z[g][0]);
        z[g][1] = wmma_bf16(a1, bb, z[g][1]);
      }
    }
    __syncthreads();  // all waves done reading h before it is overwritten

    // gates + cell update entirely in registers; write new h to LDS + output
#pragma unroll
    for (int mt = 0; mt < 2; ++mt) {
      f32x8& cc = mt ? c1 : c0;
#pragma unroll
      for (int e = 0; e < 8; ++e) {
        const float ig = sigmoid_(z[0][mt][e]);
        const float fg = sigmoid_(z[1][mt][e]);
        const float gg = tanh_   (z[2][mt][e]);
        const float og = sigmoid_(z[3][mt][e]);
        const float cn = fg * cc[e] + ig * gg;
        cc[e] = cn;
        const float hn = og * tanh_(cn);
        const int b = mt * 16 + hi * 8 + e;
        const int j = j0 + m_lane;
        hsh[b][j] = (bf16)hn;
        const size_t off = (size_t)t * BATCH * HID + (size_t)b * HID + j;
        if (write_bf) out_bf[off] = (bf16)hn;
        if (write_f)  out_f[off]  = hn;
      }
    }
    __syncthreads();  // new h visible before next step's fragment loads
  }
}

// ---------------- host side ----------------
extern "C" void kernel_launch(void* const* d_in, const int* in_sizes, int n_in,
                              void* d_out, int out_size, void* d_ws, size_t ws_size,
                              hipStream_t stream) {
  (void)in_sizes; (void)n_in; (void)out_size; (void)ws_size;
  const float* x_in = (const float*)d_in[0];  // [T,B,H]
  const float* W_ih = (const float*)d_in[1];  // [L,4H,H]
  const float* W_hh = (const float*)d_in[2];  // [L,4H,H]
  const float* b_ih = (const float*)d_in[3];  // [L,4H]
  const float* b_hh = (const float*)d_in[4];  // [L,4H]

  char* ws = (char*)d_ws;
  bf16*  xb    = (bf16*)(ws);                          // 16 MB  [T*B*H] bf16
  bf16*  hb    = (bf16*)(ws + ((size_t)16 << 20));     // 16 MB  layer-0 h output (bf16)
  bf16*  wihb  = (bf16*)(ws + ((size_t)32 << 20));     //  2 MB
  bf16*  whhb  = (bf16*)(ws + ((size_t)34 << 20));     //  2 MB
  float* xproj = (float*)(ws + ((size_t)36 << 20));    // 128 MB [T*B*4H] fp32

  const int nX = T_LEN * BATCH * HID;  // 8388608
  const int nW = G4H * HID;            // 1048576

  cvt_f32_bf16<<<(nX + 255) / 256, 256, 0, stream>>>(x_in, xb, nX);

  const bf16* xcur = xb;
  for (int l = 0; l < 2; ++l) {
    cvt_f32_bf16<<<(nW + 255) / 256, 256, 0, stream>>>(W_ih + (size_t)l * nW, wihb, nW);
    cvt_f32_bf16<<<(nW + 255) / 256, 256, 0, stream>>>(W_hh + (size_t)l * nW, whhb, nW);

    lstm_gemm_xproj<<<dim3((T_LEN * BATCH) / 128, G4H / 128), 256, 0, stream>>>(
        xcur, wihb, b_ih + (size_t)l * G4H, b_hh + (size_t)l * G4H, xproj);

    lstm_scan<<<1, 1024, 0, stream>>>(xproj, whhb, hb, (float*)d_out,
                                      l == 0 ? 1 : 0, l == 1 ? 1 : 0);
    xcur = hb;
  }
}
